// SynthesisBlock_36507222016810
// MI455X (gfx1250) — matrix operand order repaired and checked by reading, not compiled, not run on hardware
//
#include <hip/hip_runtime.h>

// ---------------- types ----------------
typedef __attribute__((ext_vector_type(16))) _Float16 v16h;
typedef __attribute__((ext_vector_type(8)))  _Float16 v8h;
typedef __attribute__((ext_vector_type(8)))  float    v8f;

// ---------------- constants ----------------
constexpr int   BATCH = 16;
constexpr int   C     = 512;     // CIN == COUT
constexpr int   RES   = 64;
constexpr int   HW    = RES * RES;        // 4096
constexpr int   UPR   = 66;               // upsampled-filtered intermediate (valid for 3x3 -> 64)
constexpr int   WTILE = 9 * 64 * 32;      // packed halves per (coTile, ciBlock)
constexpr float INV_SQRT_WDIM = 0.04419417382415922f;   // 1/sqrt(512)
constexpr float SQRT2_F       = 1.4142135623730951f;
constexpr float CLAMP_V       = 256.0f;

// =====================================================================
// StyleSplit: out[b][c] = (m1*m2 + m3) * postscale
// =====================================================================
__global__ void style_kernel(const float* __restrict__ wlat, const float* __restrict__ Wm,
                             const float* __restrict__ bv, float* __restrict__ out,
                             float postscale) {
  int idx = blockIdx.x * 256 + threadIdx.x;
  if (idx >= BATCH * C) return;
  int b = idx / C, c = idx % C;
  const float* wl = wlat + b * 512;
  float m[3];
#pragma unroll
  for (int t = 0; t < 3; ++t) {
    const float* row = Wm + (size_t)(t * C + c) * 512;
    float s = 0.f;
    for (int k = 0; k < 512; ++k) s += wl[k] * row[k];
    m[t] = s * INV_SQRT_WDIM + bv[t * C + c];
  }
  out[idx] = (m[0] * m[1] + m[2]) * postscale;
}

// wsq[co][ci] = sum_{3x3} w^2
__global__ void wsq_kernel(const float* __restrict__ w, float* __restrict__ wsq) {
  int idx = blockIdx.x * 256 + threadIdx.x;
  if (idx >= C * C) return;
  const float* p = w + (size_t)idx * 9;
  float s = 0.f;
#pragma unroll
  for (int t = 0; t < 9; ++t) s += p[t] * p[t];
  wsq[idx] = s;
}

// d[b][co] = rsqrt( sum_ci wsq[co][ci] * s[b][ci]^2 + 1e-8 )
__global__ void demod_kernel(const float* __restrict__ wsq, const float* __restrict__ s,
                             float* __restrict__ d) {
  int idx = blockIdx.x * 256 + threadIdx.x;
  if (idx >= BATCH * C) return;
  int b = idx / C, co = idx % C;
  const float* sv = s + b * C;
  const float* wr = wsq + (size_t)co * C;
  float acc = 1e-8f;
  for (int ci = 0; ci < C; ++ci) { float t = sv[ci]; acc += wr[ci] * t * t; }
  d[idx] = rsqrtf(acc);
}

// =====================================================================
// Pack weights to f16 GEMM layout: [coT(8)][cb(16)][tap(9)][co(64)][ci(32)]
// flip!=0 applies the conv0 spatial flip (tap -> 8-tap).
// =====================================================================
__global__ void wpack_kernel(const float* __restrict__ w, _Float16* __restrict__ out, int flip) {
  int idx = blockIdx.x * 256 + threadIdx.x;
  if (idx >= C * C * 9) return;
  int ci  = idx & 31;
  int co  = (idx >> 5) & 63;
  int tap = (idx / 2048) % 9;
  int cb  = (idx / (2048 * 9)) & 15;
  int coT = idx / (2048 * 9 * 16);
  int co_g = coT * 64 + co;
  int ci_g = cb * 32 + ci;
  int tap_g = flip ? (8 - tap) : tap;
  out[idx] = (_Float16)w[((size_t)co_g * C + ci_g) * 9 + tap_g];
}

// =====================================================================
// Modulate + zero-insert 2x upsample + pad(3,2) + 4x4 FIR (gain 4) -> f16 66x66
// =====================================================================
__global__ void upfir_mod_kernel(const float* __restrict__ x, const float* __restrict__ s,
                                 _Float16* __restrict__ xu) {
  const float f4[4] = {0.125f, 0.375f, 0.375f, 0.125f};
  size_t idx = (size_t)blockIdx.x * 256 + threadIdx.x;
  const size_t total = (size_t)BATCH * C * UPR * UPR;
  if (idx >= total) return;
  int j  = (int)(idx % UPR);
  int i  = (int)((idx / UPR) % UPR);
  int ci = (int)((idx / (UPR * UPR)) % C);
  int b  = (int)(idx / ((size_t)UPR * UPR * C));
  const float* xb = x + ((size_t)b * C + ci) * 32 * 32;
  float acc = 0.f;
#pragma unroll
  for (int u = 0; u < 4; ++u) {
    int rr = i + u - 3;                       // zero-inserted image coord
    if (rr < 0 || rr >= 64 || (rr & 1)) continue;
#pragma unroll
    for (int v = 0; v < 4; ++v) {
      int cc = j + v - 3;
      if (cc < 0 || cc >= 64 || (cc & 1)) continue;
      acc += 4.f * f4[u] * f4[v] * xb[(rr >> 1) * 32 + (cc >> 1)];
    }
  }
  xu[idx] = (_Float16)(acc * s[b * C + ci]);
}

// =====================================================================
// Implicit-GEMM 3x3 conv, WMMA f16->f32, 2x2 fragment blocking per wave.
// Block: 128 thr = 4 waves; tile 64 cout x 64 px (one full output row).
// Wave (wy,wx) in 2x2 grid owns a 32x32 C tile (4 accumulators).
//  MODE 0: conv0 — input xu[b][ci][66][66] (VALID), epilogue demod+bias+lrelu
//  MODE 1: residual — input xm[b][ci][64][64] pad=1, epilogue + (gamma*h+xn)*sqrt2
// =====================================================================
template <int MODE>
__global__ __launch_bounds__(128) void conv_gemm_kernel(
    const _Float16* __restrict__ xin, const _Float16* __restrict__ wpack,
    const float* __restrict__ demod, const float* __restrict__ bias,
    const float* __restrict__ gamma, const float* __restrict__ xn,
    float* __restrict__ yout) {
  const int b   = blockIdx.z;
  const int co0 = blockIdx.y * 64;
  const int h   = blockIdx.x;        // full output row
  const int tid    = threadIdx.x;
  const int lane   = tid & 31;
  const int waveId = tid >> 5;       // 0..3
  const int mo = (waveId >> 1) * 32; // cout sub-tile (2 M-frags)
  const int pw = (waveId & 1) * 32;  // pixel sub-tile (2 N-frags)

  __shared__ __align__(64) _Float16 aW[WTILE];      // [tap][co(64)][ci(32)]
  __shared__ __align__(64) _Float16 bX[3][66][32];  // [kh][col(+halo)][ci]

  v8f acc[2][2] = {};

  for (int cb = 0; cb < C / 32; ++cb) {
    const int ci0 = cb * 32;
    __syncthreads();
    // ---- stage packed weights: straight contiguous f16 copy (v8h)
    const _Float16* wsrc = wpack + ((size_t)blockIdx.y * 16 + cb) * WTILE;
    for (int id = tid * 8; id < WTILE; id += 128 * 8)
      *(v8h*)&aW[id] = *(const v8h*)&wsrc[id];
    if (cb + 1 < C / 32)
      __builtin_prefetch(wsrc + WTILE, 0, 3);       // global_prefetch next K-block
    // ---- stage inputs: 3 rows x 66 cols x 32 ci (col fastest for coalescing)
    for (int id = tid; id < 3 * 66 * 32; id += 128) {
      int kh  = id / 2112;
      int rem = id - kh * 2112;
      int ci  = rem / 66;
      int col = rem - ci * 66;
      _Float16 v;
      if (MODE == 0) {
        v = xin[(((size_t)b * C + (ci0 + ci)) * UPR + (h + kh)) * UPR + col];
      } else {
        int r = h + kh - 1, cc = col - 1;
        v = (r >= 0 && r < RES && cc >= 0 && cc < RES)
              ? xin[(((size_t)b * C + (ci0 + ci)) * RES + r) * RES + cc]
              : (_Float16)0.f;
      }
      bX[kh][col][ci] = v;
    }
    __syncthreads();
    // ---- 9 taps x 4 WMMAs over this 32-deep ci block
#pragma unroll
    for (int kh = 0; kh < 3; ++kh) {
#pragma unroll
      for (int kw = 0; kw < 3; ++kw) {
        const _Float16* abase = &aW[(kh * 3 + kw) * 2048];
        const int arow = lane & 15;
        const int akb  = (lane < 16) ? 0 : 16;  // A: lanes 0-15 K=0..15
        union { v16h v; v8h h2[2]; } af0, af1;
        af0.h2[0] = *(const v8h*)&abase[(mo + arow) * 32 + akb];
        af0.h2[1] = *(const v8h*)&abase[(mo + arow) * 32 + akb + 8];
        af1.h2[0] = *(const v8h*)&abase[(mo + 16 + arow) * 32 + akb];
        af1.h2[1] = *(const v8h*)&abase[(mo + 16 + arow) * 32 + akb + 8];
        const int bkb = (lane < 16) ? 0 : 16;   // B: lanes 0-15 K=0..15
        v16h bf0 = *(const v16h*)&bX[kh][pw + (lane & 15) + kw][bkb];
        v16h bf1 = *(const v16h*)&bX[kh][pw + 16 + (lane & 15) + kw][bkb];
        acc[0][0] = __builtin_amdgcn_wmma_f32_16x16x32_f16(false, af0.v, false, bf0,
                                                           (short)0, acc[0][0], false, false);
        acc[0][1] = __builtin_amdgcn_wmma_f32_16x16x32_f16(false, af0.v, false, bf1,
                                                           (short)0, acc[0][1], false, false);
        acc[1][0] = __builtin_amdgcn_wmma_f32_16x16x32_f16(false, af1.v, false, bf0,
                                                           (short)0, acc[1][0], false, false);
        acc[1][1] = __builtin_amdgcn_wmma_f32_16x16x32_f16(false, af1.v, false, bf1,
                                                           (short)0, acc[1][1], false, false);
      }
    }
  }
  // ---- fused epilogue (C/D layout: vgpr r -> M=r(+8), lane%16 -> N)
  const int n  = lane & 15;
  const int mb = (lane < 16) ? 0 : 8;
#pragma unroll
  for (int i = 0; i < 2; ++i) {
#pragma unroll
    for (int j = 0; j < 2; ++j) {
#pragma unroll
      for (int r = 0; r < 8; ++r) {
        int co = co0 + mo + i * 16 + mb + r;
        int wc = pw + j * 16 + n;
        float v = acc[i][j][r] * demod[b * C + co] + bias[co];
        v = (v > 0.f ? v : 0.2f * v) * SQRT2_F;           // leaky_relu * sqrt(2)
        v = fminf(fmaxf(v, -CLAMP_V), CLAMP_V);
        size_t oidx = (((size_t)b * C + co) * RES + h) * RES + wc;
        if (MODE == 1) v = (gamma[co] * v + xn[oidx]) * SQRT2_F;
        yout[oidx] = v;
      }
    }
  }
}

// =====================================================================
// GroupNorm stats: one block per (b, group of 16 ch); stats = {mu, rstd}
// =====================================================================
__global__ void gn_stats_kernel(const float* __restrict__ y, float* __restrict__ stats) {
  int bg = blockIdx.x;               // b*32 + g
  int tid = threadIdx.x;
  const float* base = y + (size_t)bg * 16 * HW;
  float s = 0.f, q = 0.f;
  for (int i = tid; i < 16 * HW; i += 256) { float v = base[i]; s += v; q += v * v; }
  __shared__ float ss[256], sq[256];
  ss[tid] = s; sq[tid] = q;
  __syncthreads();
  for (int off = 128; off > 0; off >>= 1) {
    if (tid < off) { ss[tid] += ss[tid + off]; sq[tid] += sq[tid + off]; }
    __syncthreads();
  }
  if (tid == 0) {
    float mu  = ss[0] * (1.f / 65536.f);
    float var = sq[0] * (1.f / 65536.f) - mu * mu;
    stats[2 * bg]     = mu;
    stats[2 * bg + 1] = rsqrtf(var + 1e-5f);
  }
}

// apply GN -> xn (fp32, skip path) and xm = xn * style (f16, GEMM operand)
__global__ void gn_apply_kernel(const float* __restrict__ y, const float* __restrict__ stats,
                                const float* __restrict__ gs, const float* __restrict__ gb,
                                const float* __restrict__ sr, float* __restrict__ xn,
                                _Float16* __restrict__ xm) {
  size_t idx = (size_t)blockIdx.x * 256 + threadIdx.x;
  if (idx >= (size_t)BATCH * C * HW) return;
  int c = (int)((idx >> 12) & (C - 1));
  int b = (int)(idx >> 21);
  int g = c >> 4;
  float mu = stats[2 * (b * 32 + g)], rstd = stats[2 * (b * 32 + g) + 1];
  float v = (y[idx] - mu) * rstd * gs[c] + gb[c];
  xn[idx] = v;
  xm[idx] = (_Float16)(v * sr[b * C + c]);
}

// =====================================================================
// ToRGB: img[b][c][p] = clamp( sum_co y*srgb[b][co]*wrgb[c][co] + brgb[c] )
// =====================================================================
__global__ void torgb_kernel(const float* __restrict__ y, const float* __restrict__ srgb,
                             const float* __restrict__ wrgb, const float* __restrict__ brgb,
                             float* __restrict__ img) {
  int idx = blockIdx.x * 256 + threadIdx.x;
  if (idx >= BATCH * HW) return;
  int b = idx >> 12, p = idx & (HW - 1);
  float a0 = brgb[0], a1 = brgb[1], a2 = brgb[2];
  const float* yb = y + (size_t)b * C * HW + p;
  const float* sb = srgb + b * C;
  for (int co = 0; co < C; ++co) {
    float v = yb[(size_t)co * HW] * sb[co];
    a0 += v * wrgb[co];
    a1 += v * wrgb[C + co];
    a2 += v * wrgb[2 * C + co];
  }
  a0 = fminf(fmaxf(a0, -CLAMP_V), CLAMP_V);
  a1 = fminf(fmaxf(a1, -CLAMP_V), CLAMP_V);
  a2 = fminf(fmaxf(a2, -CLAMP_V), CLAMP_V);
  img[((size_t)b * 3 + 0) * HW + p] = a0;
  img[((size_t)b * 3 + 1) * HW + p] = a1;
  img[((size_t)b * 3 + 2) * HW + p] = a2;
}

// =====================================================================
extern "C" void kernel_launch(void* const* d_in, const int* in_sizes, int n_in,
                              void* d_out, int out_size, void* d_ws, size_t ws_size,
                              hipStream_t stream) {
  (void)in_sizes; (void)n_in; (void)out_size; (void)ws_size;
  const float* x      = (const float*)d_in[0];
  const float* wlat   = (const float*)d_in[1];
  const float* w0     = (const float*)d_in[2];
  const float* b0     = (const float*)d_in[3];
  const float* a0_W   = (const float*)d_in[4];
  const float* a0_b   = (const float*)d_in[5];
  const float* wr     = (const float*)d_in[6];
  const float* br     = (const float*)d_in[7];
  const float* ar_W   = (const float*)d_in[8];
  const float* ar_b   = (const float*)d_in[9];
  const float* gn_s   = (const float*)d_in[10];
  const float* gn_b   = (const float*)d_in[11];
  const float* gamma  = (const float*)d_in[12];
  const float* wrgb   = (const float*)d_in[13];
  const float* brgb   = (const float*)d_in[14];
  const float* argb_W = (const float*)d_in[15];
  const float* argb_b = (const float*)d_in[16];

  float* yout = (float*)d_out;                                // [16][512][64][64]
  float* img  = yout + (size_t)BATCH * C * HW;                // [16][3][64][64]

  // ---- workspace carve ----
  size_t off = 0;
  auto carve = [&](size_t nbytes) -> void* {
    void* p = (char*)d_ws + off;
    off += (nbytes + 255) & ~(size_t)255;
    return p;
  };
  float* s0    = (float*)carve((size_t)BATCH * C * 4);
  float* sr0   = (float*)carve((size_t)BATCH * C * 4);
  float* sr1   = (float*)carve((size_t)BATCH * C * 4);
  float* srgb  = (float*)carve((size_t)BATCH * C * 4);
  float* d0c   = (float*)carve((size_t)BATCH * C * 4);
  float* dr0   = (float*)carve((size_t)BATCH * C * 4);
  float* dr1   = (float*)carve((size_t)BATCH * C * 4);
  float* wsq0  = (float*)carve((size_t)C * C * 4);
  float* wsqr0 = (float*)carve((size_t)C * C * 4);
  float* wsqr1 = (float*)carve((size_t)C * C * 4);
  float* stats = (float*)carve((size_t)BATCH * 32 * 2 * 4);
  _Float16* wp0  = (_Float16*)carve((size_t)C * C * 9 * 2);
  _Float16* wpr0 = (_Float16*)carve((size_t)C * C * 9 * 2);
  _Float16* wpr1 = (_Float16*)carve((size_t)C * C * 9 * 2);
  _Float16* xh   = (_Float16*)carve((size_t)BATCH * C * UPR * UPR * 2);  // reused as xm 64x64
  float* xnbuf   = (float*)carve((size_t)BATCH * C * HW * 4);

  const int TPB = 256;
  // styles
  style_kernel<<<(BATCH * C + TPB - 1) / TPB, TPB, 0, stream>>>(wlat, a0_W, a0_b, s0, 1.f);
  style_kernel<<<(BATCH * C + TPB - 1) / TPB, TPB, 0, stream>>>(wlat, ar_W, ar_b, sr0, 1.f);
  style_kernel<<<(BATCH * C + TPB - 1) / TPB, TPB, 0, stream>>>(wlat, ar_W + (size_t)3 * C * 512,
                                                                ar_b + 3 * C, sr1, 1.f);
  style_kernel<<<(BATCH * C + TPB - 1) / TPB, TPB, 0, stream>>>(wlat, argb_W, argb_b, srgb,
                                                                INV_SQRT_WDIM);
  // weight^2 sums + demod factors
  wsq_kernel<<<(C * C + TPB - 1) / TPB, TPB, 0, stream>>>(w0, wsq0);
  wsq_kernel<<<(C * C + TPB - 1) / TPB, TPB, 0, stream>>>(wr, wsqr0);
  wsq_kernel<<<(C * C + TPB - 1) / TPB, TPB, 0, stream>>>(wr + (size_t)C * C * 9, wsqr1);
  demod_kernel<<<(BATCH * C + TPB - 1) / TPB, TPB, 0, stream>>>(wsq0, s0, d0c);
  demod_kernel<<<(BATCH * C + TPB - 1) / TPB, TPB, 0, stream>>>(wsqr0, sr0, dr0);
  demod_kernel<<<(BATCH * C + TPB - 1) / TPB, TPB, 0, stream>>>(wsqr1, sr1, dr1);
  // pack weights to f16 GEMM layout (conv0 flipped)
  wpack_kernel<<<(C * C * 9 + TPB - 1) / TPB, TPB, 0, stream>>>(w0, wp0, 1);
  wpack_kernel<<<(C * C * 9 + TPB - 1) / TPB, TPB, 0, stream>>>(wr, wpr0, 0);
  wpack_kernel<<<(C * C * 9 + TPB - 1) / TPB, TPB, 0, stream>>>(wr + (size_t)C * C * 9, wpr1, 0);

  // conv0: modulate + FIR-upsample to f16, then WMMA implicit GEMM
  {
    size_t total = (size_t)BATCH * C * UPR * UPR;
    upfir_mod_kernel<<<(unsigned)((total + TPB - 1) / TPB), TPB, 0, stream>>>(x, s0, xh);
  }
  conv_gemm_kernel<0><<<dim3(RES, C / 64, BATCH), 128, 0, stream>>>(
      xh, wp0, d0c, b0, nullptr, nullptr, yout);

  // residual layers
  for (int i = 0; i < 2; ++i) {
    gn_stats_kernel<<<BATCH * 32, TPB, 0, stream>>>(yout, stats);
    gn_apply_kernel<<<(unsigned)(((size_t)BATCH * C * HW + TPB - 1) / TPB), TPB, 0, stream>>>(
        yout, stats, gn_s + i * C, gn_b + i * C, (i ? sr1 : sr0), xnbuf, xh);
    conv_gemm_kernel<1><<<dim3(RES, C / 64, BATCH), 128, 0, stream>>>(
        xh, (i ? wpr1 : wpr0), (i ? dr1 : dr0), br + i * C, gamma + i * C, xnbuf, yout);
  }

  // ToRGB
  torgb_kernel<<<(BATCH * HW + TPB - 1) / TPB, TPB, 0, stream>>>(yout, srgb, wrgb, brgb, img);
}
